// MoELayer_4002909520313
// MI455X (gfx1250) — matrix operand order repaired and checked by reading, not compiled, not run on hardware
//
#include <hip/hip_runtime.h>
#include <cstdint>
#include <cstddef>

typedef __bf16 bf16_t;
typedef __attribute__((ext_vector_type(8)))  float  v8f;
typedef __attribute__((ext_vector_type(8)))  bf16_t v8bf;
typedef __attribute__((ext_vector_type(16))) bf16_t v16bf;

static constexpr int T = 8192;   // B*S tokens
static constexpr int D = 1024;   // model dim
static constexpr int H = 4096;   // hidden dim (4*D)
static constexpr int E = 8;      // experts
static constexpr int M_TILE = 32;   // tokens per workgroup
static constexpr int HC = 128;      // hidden-chunk width (fused, never hits HBM)

__device__ __forceinline__ v16bf cat8(v8bf a, v8bf b) {
  return __builtin_shufflevector(a, b, 0,1,2,3,4,5,6,7,8,9,10,11,12,13,14,15);
}

// ---------------- x -> bf16 ----------------
__global__ __launch_bounds__(256) void cvt_x_kernel(const float* __restrict__ x,
                                                    bf16_t* __restrict__ xb, int n) {
  int i = blockIdx.x * blockDim.x + threadIdx.x;
  int stride = gridDim.x * blockDim.x;
  for (; i < n; i += stride) xb[i] = (bf16_t)x[i];
}

// ---- weights: out[e][c][r] = (bf16) in[e][r][c]  (transpose to N-major) ----
__global__ __launch_bounds__(256) void transpose_cvt_kernel(const float* __restrict__ in,
                                                            bf16_t* __restrict__ out,
                                                            int R, int C) {
  __shared__ float tile[32][33];
  const int e  = blockIdx.z;
  const int r0 = blockIdx.x * 32;
  const int c0 = blockIdx.y * 32;
  const int tx = threadIdx.x, ty = threadIdx.y;
  const float* ip = in  + (size_t)e * R * C;
  bf16_t*      op = out + (size_t)e * R * C;
  #pragma unroll
  for (int i = ty; i < 32; i += 8)
    tile[i][tx] = ip[(size_t)(r0 + i) * C + (c0 + tx)];
  __syncthreads();
  #pragma unroll
  for (int i = ty; i < 32; i += 8)
    op[(size_t)(c0 + i) * R + (r0 + tx)] = (bf16_t)tile[tx][i];
}

// ---------------- gating: logits, top-2, softmax, route ----------------
__global__ __launch_bounds__(256) void gate_route_kernel(const float* __restrict__ x,
                                                         const float* __restrict__ wg,
                                                         int*   __restrict__ counts,
                                                         int*   __restrict__ toks,
                                                         float* __restrict__ gates) {
  const int wave = threadIdx.x >> 5;
  const int lane = threadIdx.x & 31;
  const int t = blockIdx.x * 8 + wave;   // one token per wave32
  const float* xr = x + (size_t)t * D;
  float acc[E];
  #pragma unroll
  for (int e = 0; e < E; ++e) acc[e] = 0.0f;
  for (int d = lane; d < D; d += 32) {
    float xv = xr[d];
    const float* wr = wg + (size_t)d * E;
    #pragma unroll
    for (int e = 0; e < E; ++e) acc[e] += xv * wr[e];
  }
  #pragma unroll
  for (int e = 0; e < E; ++e)
    #pragma unroll
    for (int off = 16; off; off >>= 1) acc[e] += __shfl_xor(acc[e], off, 32);

  if (lane == 0) {
    int i0 = 0; float v0 = acc[0];
    #pragma unroll
    for (int e = 1; e < E; ++e) if (acc[e] > v0) { v0 = acc[e]; i0 = e; }
    int i1 = (i0 == 0) ? 1 : 0; float v1 = acc[i1];
    #pragma unroll
    for (int e = 0; e < E; ++e) if (e != i0 && acc[e] > v1) { v1 = acc[e]; i1 = e; }
    float ex = __expf(v1 - v0);            // stable 2-way softmax
    float g0 = 1.0f / (1.0f + ex);
    float g1 = ex   / (1.0f + ex);
    int p0 = atomicAdd(&counts[i0], 1);
    toks[i0 * T + p0] = t;  gates[i0 * T + p0] = g0;
    int p1 = atomicAdd(&counts[i1], 1);
    toks[i1 * T + p1] = t;  gates[i1 * T + p1] = g1;
  }
}

// ---------------- fused expert FFN: y = relu(x @ W1) @ W2, gated scatter ----------------
__global__ __launch_bounds__(512, 1) void moe_ffn_kernel(
    const bf16_t* __restrict__ xb,      // [T][D]    bf16
    const bf16_t* __restrict__ w1t,     // [E][H][D] bf16, n-major
    const bf16_t* __restrict__ w2t,     // [E][D][H] bf16, n-major
    const int*    __restrict__ counts,
    const int*    __restrict__ toks,
    const float*  __restrict__ gates,
    float* __restrict__ out) {
  const int e    = blockIdx.y;
  const int cnt  = counts[e];
  const int row0 = blockIdx.x * M_TILE;
  if (row0 >= cnt) return;              // uniform early exit (before any barrier/WMMA)

  __shared__ bf16_t xs[M_TILE][D];      // 64 KB: gathered activations
  __shared__ bf16_t hs[M_TILE][HC];     // 8 KB: relu(h) chunk
  __shared__ int    stok[M_TILE];
  __shared__ float  sg[M_TILE];

  const int tid = threadIdx.x;
  if (tid < M_TILE) {
    int r = row0 + tid;
    bool v = (r < cnt);
    stok[tid] = v ? toks[e * T + r]  : 0;
    sg[tid]   = v ? gates[e * T + r] : 0.0f;
  }
  __syncthreads();
  { // gather 32 token rows of x (bf16) into LDS: 16 threads/row, 8x16B each
    int m = tid >> 4;
    int c = (tid & 15) * 64;
    const v8bf* src = (const v8bf*)(xb + (size_t)stok[m] * D + c);
    v8bf* dst = (v8bf*)&xs[m][c];
    #pragma unroll
    for (int j = 0; j < 8; ++j) dst[j] = src[j];
  }
  __syncthreads();

  const int wave  = tid >> 5;        // 16 waves
  const int lane  = tid & 31;
  const int l16   = lane & 15;
  const int lhalf = lane >> 4;
  const int mrow  = (wave & 1) * 16; // token sub-tile
  const int nspan = wave >> 1;       // 0..7

  const bf16_t* w1e = w1t + (size_t)e * H * D;
  const bf16_t* w2e = w2t + (size_t)e * D * H;

  v8f yc[8] = {};                    // 8 x (16x16) f32 accum tiles per wave

  for (int hc = 0; hc < H; hc += HC) {
    // ---- h tile: rows [mrow,mrow+16), hidden cols hc + nspan*16 + [0,16) ----
    v8f ha = {};
    const bf16_t* w1p = w1e + (size_t)(hc + nspan * 16 + l16) * D + lhalf * 16;
    #pragma unroll 4
    for (int k0 = 0; k0 < D; k0 += 32) {
      const v8bf* ap = (const v8bf*)&xs[mrow + l16][k0 + lhalf * 16]; // A layout
      v16bf a = cat8(ap[0], ap[1]);                                  // K: 16 contiguous per half
      v16bf b = *(const v16bf*)(w1p + k0);                           // B layout: 16 contiguous K halves
      ha = __builtin_amdgcn_wmma_f32_16x16x32_bf16(false, a, false, b, (short)0, ha, false, false);
    }
    __syncthreads();                 // previous chunk's y-GEMM done reading hs
    #pragma unroll
    for (int r = 0; r < 8; ++r) {    // C layout: M = r + 8*lhalf, N = l16
      float v = ha[r];
      hs[mrow + r + 8 * lhalf][nspan * 16 + l16] = (bf16_t)(v > 0.0f ? v : 0.0f);
    }
    __syncthreads();
    // ---- y += hs @ W2[hc:hc+HC, :], wave covers n in [nspan*128, +128) ----
    #pragma unroll
    for (int k0 = 0; k0 < HC; k0 += 32) {
      const v8bf* hp = (const v8bf*)&hs[mrow + l16][k0 + lhalf * 16];
      v16bf a = cat8(hp[0], hp[1]);
      #pragma unroll
      for (int nt = 0; nt < 8; ++nt) {
        const bf16_t* w2p = w2e + (size_t)(nspan * 128 + nt * 16 + l16) * H
                                + (hc + k0 + lhalf * 16);
        v16bf b = *(const v16bf*)w2p;
        yc[nt] = __builtin_amdgcn_wmma_f32_16x16x32_bf16(false, a, false, b, (short)0, yc[nt], false, false);
      }
    }
  }

  // gated scatter-accumulate (each token hit by exactly 2 experts -> f32 atomics)
  #pragma unroll
  for (int nt = 0; nt < 8; ++nt) {
    #pragma unroll
    for (int r = 0; r < 8; ++r) {
      int m = mrow + r + 8 * lhalf;
      float g = sg[m];
      if (g != 0.0f) {
        int n = nspan * 128 + nt * 16 + l16;
        atomicAdd(&out[(size_t)stok[m] * D + n], g * yc[nt][r]);
      }
    }
  }
}

extern "C" void kernel_launch(void* const* d_in, const int* in_sizes, int n_in,
                              void* d_out, int out_size, void* d_ws, size_t ws_size,
                              hipStream_t stream) {
  (void)in_sizes; (void)n_in; (void)ws_size;
  const float* x  = (const float*)d_in[0];
  const float* wg = (const float*)d_in[1];
  const float* w1 = (const float*)d_in[2];
  const float* w2 = (const float*)d_in[3];
  float* out = (float*)d_out;

  char* ws = (char*)d_ws;
  size_t off = 0;
  bf16_t* w1t = (bf16_t*)(ws + off); off += (size_t)E * H * D * sizeof(bf16_t); // 67 MB
  bf16_t* w2t = (bf16_t*)(ws + off); off += (size_t)E * H * D * sizeof(bf16_t); // 67 MB
  bf16_t* xb  = (bf16_t*)(ws + off); off += (size_t)T * D * sizeof(bf16_t);     // 16.8 MB
  int*    toks   = (int*)  (ws + off); off += (size_t)E * T * sizeof(int);
  float*  gatesb = (float*)(ws + off); off += (size_t)E * T * sizeof(float);
  int*    counts = (int*)  (ws + off); off += 256;

  hipMemsetAsync(counts, 0, E * sizeof(int), stream);
  hipMemsetAsync(out, 0, (size_t)out_size * sizeof(float), stream);

  cvt_x_kernel<<<dim3(2048), dim3(256), 0, stream>>>(x, xb, T * D);
  transpose_cvt_kernel<<<dim3(D / 32, H / 32, E), dim3(32, 8), 0, stream>>>(w1, w1t, D, H);
  transpose_cvt_kernel<<<dim3(H / 32, D / 32, E), dim3(32, 8), 0, stream>>>(w2, w2t, H, D);
  gate_route_kernel<<<dim3(T / 8), dim3(256), 0, stream>>>(x, wg, counts, toks, gatesb);
  moe_ffn_kernel<<<dim3(T / M_TILE, E), dim3(512), 0, stream>>>(xb, w1t, w2t, counts, toks, gatesb, out);
}
